// ItemAgg_21354577396101
// MI455X (gfx1250) — matrix operand order repaired and checked by reading, not compiled, not run on hardware
//
#include <hip/hip_runtime.h>
#include <hip/hip_bf16.h>

typedef _Float16 v16h __attribute__((ext_vector_type(16)));
typedef _Float16 v8h  __attribute__((ext_vector_type(8)));
typedef _Float16 v4h  __attribute__((ext_vector_type(4)));
typedef float    v8f  __attribute__((ext_vector_type(8)));

#define WAVES_PER_BLOCK 8
#define TS  264   // edge tile row stride (halves): 256 cols + 8 pad (bank spread)
#define TS2 136   // out tile row stride (halves): 128 cols + 8 pad

// ---------------------------------------------------------------------------
// A-fragment load for V_WMMA_F32_16X16X32_F16 from a row-major f16 tile in LDS.
// Lanes 0-15: row M=lane, halves K = kb*32 + {0..7, 16..23}
// Lanes 16-31: row M=lane-16, halves K = kb*32 + {8..15, 24..31}
// Two 16B ds_read_b128 per fragment.
// ---------------------------------------------------------------------------
__device__ __forceinline__ v16h load_a_frag(const _Float16* tile, int stride, int lane, int kb) {
  const int s = (lane & 16) ? 8 : 0;
  const int m = lane & 15;
  const _Float16* p = tile + m * stride + kb * 32 + s;
  v8h lo = *(const v8h*)p;
  v8h hi = *(const v8h*)(p + 16);
  return __builtin_shufflevector(lo, hi, 0,1,2,3,4,5,6,7,8,9,10,11,12,13,14,15);
}

// ---------------------------------------------------------------------------
// Pack a row-major fp32 weight [K][128] into f16 WMMA B-fragments.
// Fragment(kb,nb) = 32x16 tile, stored lane-major: 512 halves / frag,
// lane holds its 16 halves contiguously (one v16h = its whole B operand).
// B layout (16-bit, 32x16): lane<16 -> N=lane, K = {0..7, 16..23};
//                           lane>=16 -> N=lane-16, K = {8..15, 24..31}.
// ---------------------------------------------------------------------------
__global__ void pack_weight_f16(const float* __restrict__ W, _Float16* __restrict__ out, int K) {
  const int total = (K / 32) * 8 * 512;
  for (int o = blockIdx.x * blockDim.x + threadIdx.x; o < total; o += gridDim.x * blockDim.x) {
    const int frag   = o >> 9;
    const int within = o & 511;
    const int lane   = within >> 4;
    const int t      = within & 15;
    const int v      = t >> 1;
    const int hh     = t & 1;
    const int kb     = frag >> 3;
    const int nb     = frag & 7;
    const int n      = nb * 16 + (lane & 15);
    const int k      = kb * 32 + ((lane & 16) ? 8 : 0) + 2 * v + hh + ((t & 8) ? 8 : 0);
    out[o] = (_Float16)W[k * 128 + n];
  }
}

__global__ void zero_kernel(unsigned* __restrict__ p, long n) {
  for (long i = (long)blockIdx.x * blockDim.x + threadIdx.x; i < n; i += (long)gridDim.x * blockDim.x)
    p[i] = 0u;
}

// Order-preserving float <-> uint for atomicMax-based segment max
__device__ __forceinline__ unsigned fkey(float f) {
  unsigned b = __float_as_uint(f);
  return (b & 0x80000000u) ? ~b : (b | 0x80000000u);
}
__device__ __forceinline__ float kinv(unsigned k) {
  unsigned b = (k & 0x80000000u) ? (k & 0x7fffffffu) : ~k;
  return __uint_as_float(b);
}

// ---------------------------------------------------------------------------
// 16x128 tile GEMM accumulation with software-pipelined fragment loads.
// acc starts at zero so the first WMMA of each chain gets an inline SRC2=0;
// bias is folded into the activation epilogue instead.
// B (and next-kb A) fragments are fetched one WMMA ahead so the s_wait_dscnt
// in front of each WMMA covers a load issued a full WMMA earlier.
// ---------------------------------------------------------------------------
#define LAYER_MM(KB, WLDS, STRIDE)                                                          \
  {                                                                                         \
    _Pragma("unroll") for (int nb = 0; nb < 8; ++nb)                                        \
      _Pragma("unroll") for (int j = 0; j < 8; ++j) acc[nb][j] = 0.f;                       \
    v16h afrag = load_a_frag(tile, (STRIDE), lane, 0);                                      \
    v16h bcur  = *(const v16h*)((WLDS) + (lane << 4));                                      \
    _Pragma("unroll") for (int kb = 0; kb < (KB); ++kb) {                                   \
      v16h anext = afrag;                                                                   \
      if (kb + 1 < (KB)) anext = load_a_frag(tile, (STRIDE), lane, kb + 1);                 \
      _Pragma("unroll") for (int nb = 0; nb < 8; ++nb) {                                    \
        v16h bnext = bcur;                                                                  \
        if (!(kb == (KB) - 1 && nb == 7))                                                   \
          bnext = *(const v16h*)((WLDS) + ((kb * 8 + nb + 1) << 9) + (lane << 4));          \
        acc[nb] = __builtin_amdgcn_wmma_f32_16x16x32_f16(false, afrag, false, bcur,         \
                                                         (short)0, acc[nb], false, false);  \
        bcur = bnext;                                                                       \
      }                                                                                     \
      afrag = anext;                                                                        \
    }                                                                                       \
  }

#define LRELU_BIAS(BIASOFF)                                                                 \
  { _Pragma("unroll") for (int nb = 0; nb < 8; ++nb) {                                      \
      const float bv = sB[(BIASOFF) + nb * 16 + nbase];                                     \
      _Pragma("unroll") for (int j = 0; j < 8; ++j) {                                       \
        float v = acc[nb][j] + bv; acc[nb][j] = v > 0.f ? v : 0.01f * v; } } }

#define RELU_BIAS(BIASOFF)                                                                  \
  { _Pragma("unroll") for (int nb = 0; nb < 8; ++nb) {                                      \
      const float bv = sB[(BIASOFF) + nb * 16 + nbase];                                     \
      _Pragma("unroll") for (int j = 0; j < 8; ++j)                                         \
        acc[nb][j] = fmaxf(acc[nb][j] + bv, 0.f); } }

// Write C-fragments (f32) back into the row-major f16 tile at column COL0.
// C layout: lane<16 -> N=lane, M=r ; lane>=16 -> N=lane-16, M=r+8.
#define WRITEBACK(COL0)                                                                     \
  {                                                                                         \
    _Pragma("unroll") for (int nb = 0; nb < 8; ++nb) {                                      \
      const int n = (COL0) + nb * 16 + nbase;                                               \
      _Pragma("unroll") for (int r = 0; r < 8; ++r)                                         \
        tile[(mbase + r) * TS + n] = (_Float16)acc[nb][r];                                  \
    }                                                                                       \
    asm volatile("s_wait_dscnt 0" ::: "memory");                                            \
  }

// ---------------------------------------------------------------------------
// Fused 4-layer edge MLP. One wave = one 16-edge tile, 192 WMMAs per tile.
// Weights live in LDS (~197KB), per-wave row tile 16x264 halves.
// ---------------------------------------------------------------------------
__global__ __launch_bounds__(256) void edge_mlp_kernel(
    const int* __restrict__ gi, const int* __restrict__ gu, const int* __restrict__ gr,
    const float* __restrict__ user_feat, const float* __restrict__ item_feat,
    const float* __restrict__ rating_feat,
    const _Float16* __restrict__ W1p, const _Float16* __restrict__ A1p,
    const _Float16* __restrict__ W2p, const _Float16* __restrict__ A2p,
    const float* __restrict__ b1, const float* __restrict__ b2,
    const float* __restrict__ ab1, const float* __restrict__ ab2,
    const float* __restrict__ A3, const float* __restrict__ ab3,
    _Float16* __restrict__ xia_out, float* __restrict__ w_out, int E)
{
  extern __shared__ char smem[];
  _Float16* sW1 = (_Float16*)smem;       // 32768 halves
  _Float16* sA1 = sW1 + 32768;           // 32768
  _Float16* sW2 = sA1 + 32768;           // 16384
  _Float16* sA2 = sW2 + 16384;           // 16384
  float*    sB  = (float*)(sA2 + 16384); // b1|b2|ab1|ab2|A3 : 640 floats
  _Float16* tiles = (_Float16*)(sB + 640);

  const int tid = threadIdx.x;
  for (int j = tid; j < 4096; j += 256) ((uint4*)sW1)[j] = ((const uint4*)W1p)[j];
  for (int j = tid; j < 4096; j += 256) ((uint4*)sA1)[j] = ((const uint4*)A1p)[j];
  for (int j = tid; j < 2048; j += 256) ((uint4*)sW2)[j] = ((const uint4*)W2p)[j];
  for (int j = tid; j < 2048; j += 256) ((uint4*)sA2)[j] = ((const uint4*)A2p)[j];
  if (tid < 128) {
    sB[tid] = b1[tid]; sB[128 + tid] = b2[tid]; sB[256 + tid] = ab1[tid];
    sB[384 + tid] = ab2[tid]; sB[512 + tid] = A3[tid];
  }
  __syncthreads();

  const int lane  = tid & 31;
  const int wave  = tid >> 5;
  const int nbase = lane & 15;
  const int mbase = (lane & 16) ? 8 : 0;
  _Float16* tile  = tiles + wave * (16 * TS);
  const float ab3v = ab3[0];

  const int ntiles  = (E + 15) >> 4;
  const int gw      = blockIdx.x * WAVES_PER_BLOCK + wave;
  const int gstride = gridDim.x * WAVES_PER_BLOCK;

  for (int t = gw; t < ntiles; t += gstride) {
    const int e0 = t << 4;
    v8f acc[8];

    // gather x = [item_emb || rating_emb], fp32 -> f16 into LDS rows
    for (int m = 0; m < 16; ++m) {
      int e = e0 + m; if (e >= E) e = E - 1;
      const int ii = gi[e];
      const int rr = gr[e];
      float4 av = ((const float4*)item_feat)[ii * 32 + lane];
      float4 bv = ((const float4*)rating_feat)[rr * 32 + lane];
      v4h pa = { (_Float16)av.x, (_Float16)av.y, (_Float16)av.z, (_Float16)av.w };
      v4h pb = { (_Float16)bv.x, (_Float16)bv.y, (_Float16)bv.z, (_Float16)bv.w };
      *(v4h*)(tile + m * TS + lane * 4)       = pa;
      *(v4h*)(tile + m * TS + 128 + lane * 4) = pb;
    }
    asm volatile("s_wait_dscnt 0" ::: "memory");

    LAYER_MM(8, sW1, TS)     // t1 = x @ W1
    LRELU_BIAS(0)            // lrelu(t1 + b1)
    WRITEBACK(0)             // t1 -> tile cols [0,128)

    LAYER_MM(4, sW2, TS)     // x_ia = t1 @ W2
    LRELU_BIAS(128)

    // emit x_ia (f16) to global for the aggregation pass
    #pragma unroll
    for (int nb = 0; nb < 8; ++nb) {
      const int n = nb * 16 + nbase;
      #pragma unroll
      for (int r = 0; r < 8; ++r) {
        const int e = e0 + mbase + r;
        if (e < E) xia_out[(size_t)e * 128 + n] = (_Float16)acc[nb][r];
      }
    }
    // gather u_emb into cols [128,256)
    for (int m = 0; m < 16; ++m) {
      int e = e0 + m; if (e >= E) e = E - 1;
      const int uu = gu[e];
      float4 av = ((const float4*)user_feat)[uu * 32 + lane];
      v4h pa = { (_Float16)av.x, (_Float16)av.y, (_Float16)av.z, (_Float16)av.w };
      *(v4h*)(tile + m * TS + 128 + lane * 4) = pa;
    }
    WRITEBACK(0)             // x_ia -> tile cols [0,128); waits on dscnt

    LAYER_MM(8, sA1, TS)     // h1 = [x_ia||u] @ A1
    RELU_BIAS(256)
    WRITEBACK(0)

    LAYER_MM(4, sA2, TS)     // h2 = h1 @ A2
    RELU_BIAS(384)

    // w = h2 @ A3 + ab3 : per-row dot over 128 cols (8 nb-chunks x 16 lanes)
    float s[8];
    #pragma unroll
    for (int r = 0; r < 8; ++r) {
      float a = 0.f;
      #pragma unroll
      for (int nb = 0; nb < 8; ++nb) a = fmaf(acc[nb][r], sB[512 + nb * 16 + nbase], a);
      s[r] = a;
    }
    #pragma unroll
    for (int r = 0; r < 8; ++r) {
      s[r] += __shfl_xor(s[r], 1);
      s[r] += __shfl_xor(s[r], 2);
      s[r] += __shfl_xor(s[r], 4);
      s[r] += __shfl_xor(s[r], 8);
    }
    if (nbase == 0) {
      #pragma unroll
      for (int r = 0; r < 8; ++r) {
        const int e = e0 + mbase + r;
        if (e < E) w_out[e] = s[r] + ab3v;
      }
    }
  }
}

__global__ void edge_max_kernel(const float* __restrict__ w, const int* __restrict__ gu,
                                unsigned* __restrict__ mkey, int E) {
  int e = blockIdx.x * blockDim.x + threadIdx.x;
  if (e < E) atomicMax(&mkey[gu[e]], fkey(w[e]));
}

__global__ void edge_exp_kernel(const float* __restrict__ w, const int* __restrict__ gu,
                                const unsigned* __restrict__ mkey,
                                float* __restrict__ evals, float* __restrict__ denom, int E) {
  int e = blockIdx.x * blockDim.x + threadIdx.x;
  if (e >= E) return;
  const int u = gu[e];
  const float m  = kinv(mkey[u]);
  const float ev = __expf(w[e] - m);
  evals[e] = ev;
  __hip_atomic_fetch_add(&denom[u], ev, __ATOMIC_RELAXED, __HIP_MEMORY_SCOPE_AGENT);
}

// h[u] += x_ia[e] * softmax coef ; thread = (edge, 4 columns)
__global__ void edge_agg_kernel(const _Float16* __restrict__ xia,
                                const float* __restrict__ evals, const float* __restrict__ denom,
                                const int* __restrict__ gu, float* __restrict__ hbuf, int E) {
  long g = (long)blockIdx.x * blockDim.x + threadIdx.x;
  int e = (int)(g >> 5);
  if (e >= E) return;
  int c = ((int)g & 31) * 4;
  const int u = gu[e];
  const float coef = evals[e] / denom[u];
  v4h x = *(const v4h*)(xia + (size_t)e * 128 + c);
  float* dst = hbuf + (size_t)u * 128 + c;
  #pragma unroll
  for (int j = 0; j < 4; ++j)
    __hip_atomic_fetch_add(&dst[j], (float)x[j] * coef, __ATOMIC_RELAXED, __HIP_MEMORY_SCOPE_AGENT);
}

// out = h @ Wo + bo via WMMA (16-user tiles)
__global__ __launch_bounds__(256) void out_gemm_kernel(
    const float* __restrict__ hbuf, const _Float16* __restrict__ Wop,
    const float* __restrict__ bo, float* __restrict__ out, int U)
{
  extern __shared__ char smem[];
  _Float16* sW  = (_Float16*)smem;       // 16384 halves
  float*    sbo = (float*)(sW + 16384);
  _Float16* tiles = (_Float16*)(sbo + 128);

  const int tid = threadIdx.x;
  for (int j = tid; j < 2048; j += 256) ((uint4*)sW)[j] = ((const uint4*)Wop)[j];
  if (tid < 128) sbo[tid] = bo[tid];
  __syncthreads();

  const int lane  = tid & 31;
  const int wave  = tid >> 5;
  const int nbase = lane & 15;
  const int mbase = (lane & 16) ? 8 : 0;
  _Float16* tile  = tiles + wave * (16 * TS2);

  const int ntiles  = (U + 15) >> 4;
  const int gw      = blockIdx.x * WAVES_PER_BLOCK + wave;
  const int gstride = gridDim.x * WAVES_PER_BLOCK;

  for (int t = gw; t < ntiles; t += gstride) {
    const int u0 = t << 4;
    for (int m = 0; m < 16; ++m) {
      int row = u0 + m; if (row >= U) row = U - 1;
      float4 v = ((const float4*)hbuf)[row * 32 + lane];
      v4h p = { (_Float16)v.x, (_Float16)v.y, (_Float16)v.z, (_Float16)v.w };
      *(v4h*)(tile + m * TS2 + lane * 4) = p;
    }
    asm volatile("s_wait_dscnt 0" ::: "memory");

    v8f acc[8];
    LAYER_MM(4, sW, TS2)     // h @ Wo (inline-0 C on first chain, pipelined)

    #pragma unroll
    for (int nb = 0; nb < 8; ++nb) {
      const int n = nb * 16 + nbase;
      const float bv = sbo[n];
      #pragma unroll
      for (int r = 0; r < 8; ++r) {
        const int row = u0 + mbase + r;
        if (row < U) out[(size_t)row * 128 + n] = acc[nb][r] + bv;
      }
    }
  }
}

extern "C" void kernel_launch(void* const* d_in, const int* in_sizes, int n_in,
                              void* d_out, int out_size, void* d_ws, size_t ws_size,
                              hipStream_t stream) {
  (void)n_in; (void)out_size; (void)ws_size;
  const int*   gi = (const int*)d_in[0];
  const int*   gu = (const int*)d_in[1];
  const int*   gr = (const int*)d_in[2];
  const float* user_feat   = (const float*)d_in[3];
  const float* item_feat   = (const float*)d_in[4];
  const float* rating_feat = (const float*)d_in[5];
  const float* W1  = (const float*)d_in[6];
  const float* b1  = (const float*)d_in[7];
  const float* W2  = (const float*)d_in[8];
  const float* b2  = (const float*)d_in[9];
  const float* A1  = (const float*)d_in[10];
  const float* ab1 = (const float*)d_in[11];
  const float* A2  = (const float*)d_in[12];
  const float* ab2 = (const float*)d_in[13];
  const float* A3  = (const float*)d_in[14];
  const float* ab3 = (const float*)d_in[15];
  const float* Wo  = (const float*)d_in[16];
  const float* bo  = (const float*)d_in[17];

  const int E = in_sizes[0];
  const int U = in_sizes[3] / 128;

  char* ws = (char*)d_ws;
  size_t off = 0;
  auto take = [&](size_t bytes) -> char* {
    char* p = ws + off;
    off += (bytes + 255) & ~(size_t)255;
    return p;
  };
  _Float16* W1p  = (_Float16*)take((size_t)32768 * 2);
  _Float16* A1p  = (_Float16*)take((size_t)32768 * 2);
  _Float16* W2p  = (_Float16*)take((size_t)16384 * 2);
  _Float16* A2p  = (_Float16*)take((size_t)16384 * 2);
  _Float16* Wop  = (_Float16*)take((size_t)16384 * 2);
  float*    wbuf  = (float*)take((size_t)E * 4);
  float*    evals = (float*)take((size_t)E * 4);
  unsigned* mkey  = (unsigned*)take((size_t)U * 4);
  float*    denom = (float*)take((size_t)U * 4);
  float*    hbuf  = (float*)take((size_t)U * 512);
  _Float16* xia   = (_Float16*)take((size_t)E * 256);

  pack_weight_f16<<<64, 256, 0, stream>>>(W1, W1p, 256);
  pack_weight_f16<<<64, 256, 0, stream>>>(A1, A1p, 256);
  pack_weight_f16<<<64, 256, 0, stream>>>(W2, W2p, 128);
  pack_weight_f16<<<64, 256, 0, stream>>>(A2, A2p, 128);
  pack_weight_f16<<<64, 256, 0, stream>>>(Wo, Wop, 128);

  zero_kernel<<<256, 256, 0, stream>>>(mkey, (long)U);
  zero_kernel<<<256, 256, 0, stream>>>((unsigned*)denom, (long)U);
  zero_kernel<<<2048, 256, 0, stream>>>((unsigned*)hbuf, (long)U * 128);

  const size_t smem1 = (size_t)(32768 + 32768 + 16384 + 16384) * 2 + 640 * 4
                     + (size_t)WAVES_PER_BLOCK * 16 * TS * 2;
  edge_mlp_kernel<<<512, 256, smem1, stream>>>(
      gi, gu, gr, user_feat, item_feat, rating_feat,
      W1p, A1p, W2p, A2p, b1, b2, ab1, ab2, A3, ab3, xia, wbuf, E);

  edge_max_kernel<<<(E + 255) / 256, 256, 0, stream>>>(wbuf, gu, mkey, E);
  edge_exp_kernel<<<(E + 255) / 256, 256, 0, stream>>>(wbuf, gu, mkey, evals, denom, E);

  const long na = (long)E * 32;
  edge_agg_kernel<<<(int)((na + 255) / 256), 256, 0, stream>>>(xia, evals, denom, gu, hbuf, E);

  const size_t smem2 = (size_t)16384 * 2 + 128 * 4 + (size_t)WAVES_PER_BLOCK * 16 * TS2 * 2;
  out_gemm_kernel<<<512, 256, smem2, stream>>>(hbuf, Wop, bo, (float*)d_out, U);
}
